// MambaLayer_14224931685034
// MI455X (gfx1250) — compile-verified, matrix operand order checked by reading, not run on hardware
//
#include <hip/hip_runtime.h>
#include <stdint.h>

#define D_MODEL   768
#define D_STATE   16
#define D_CONV    4
#define D_INNER   1536
#define DT_RANK   48
#define DT_RANK_P 64
#define B_SZ      2
#define L_SEQ     2048
#define N_TOK     (B_SZ * L_SEQ)          // 4096 tokens
#define XPROJ_OUT (DT_RANK + 2 * D_STATE) // 80 logical
#define XP_LD     128                      // padded to one full block tile

typedef __attribute__((ext_vector_type(16))) __bf16 v16bf;
typedef __attribute__((ext_vector_type(8)))  float  v8f;
typedef __attribute__((ext_vector_type(4)))  float  f32x4;

union FragBF { v16bf v; f32x4 f[2]; };

__device__ __forceinline__ unsigned short f32_to_bf16(float f) {
  unsigned int u = __float_as_uint(f);
  u += 0x7FFFu + ((u >> 16) & 1u);   // round-to-nearest-even
  return (unsigned short)(u >> 16);
}

// ---------------------------------------------------------------------------
// fp32 -> bf16 convert with row AND column zero-padding.
// out is [out_rows, out_ld]; in is [used_rows, in_ld] (used_cols consumed).
// ---------------------------------------------------------------------------
__global__ void pad_convert_bf16(const float* __restrict__ in,
                                 unsigned short* __restrict__ out,
                                 int out_rows, int used_rows, int used_cols,
                                 int in_ld, int out_ld) {
  int idx = blockIdx.x * blockDim.x + threadIdx.x;
  int total = out_rows * out_ld;
  if (idx >= total) return;
  int r = idx / out_ld;
  int c = idx - r * out_ld;
  float v = (r < used_rows && c < used_cols) ? in[(size_t)r * in_ld + c] : 0.0f;
  out[idx] = f32_to_bf16(v);
}

// ---------------------------------------------------------------------------
// LayerNorm over D_MODEL=768, one 256-thread block per token, bf16 output.
// ---------------------------------------------------------------------------
__global__ void layernorm_kernel(const float* __restrict__ x,
                                 const float* __restrict__ w,
                                 const float* __restrict__ b,
                                 unsigned short* __restrict__ out) {
  __shared__ float red[256];
  int tok = blockIdx.x;
  int tid = threadIdx.x;
  const float* xr = x + (size_t)tok * D_MODEL;
  float v0 = xr[tid], v1 = xr[tid + 256], v2 = xr[tid + 512];
  red[tid] = v0 + v1 + v2;
  __syncthreads();
  for (int s = 128; s > 0; s >>= 1) {
    if (tid < s) red[tid] += red[tid + s];
    __syncthreads();
  }
  float mu = red[0] * (1.0f / D_MODEL);
  __syncthreads();
  float d0 = v0 - mu, d1 = v1 - mu, d2 = v2 - mu;
  red[tid] = d0 * d0 + d1 * d1 + d2 * d2;
  __syncthreads();
  for (int s = 128; s > 0; s >>= 1) {
    if (tid < s) red[tid] += red[tid + s];
    __syncthreads();
  }
  float rs = rsqrtf(red[0] * (1.0f / D_MODEL) + 1e-5f);
  unsigned short* orow = out + (size_t)tok * D_MODEL;
  orow[tid]       = f32_to_bf16(d0 * rs * w[tid]       + b[tid]);
  orow[tid + 256] = f32_to_bf16(d1 * rs * w[tid + 256] + b[tid + 256]);
  orow[tid + 512] = f32_to_bf16(d2 * rs * w[tid + 512] + b[tid + 512]);
}

// ---------------------------------------------------------------------------
// bf16 WMMA GEMM: C[M,N] = A[M,K] * B[N,K]^T  (both row-major, K innermost).
// Block = 256 threads = 8 waves arranged 4(M) x 2(N); each wave owns a
// 16x64 strip (four 16x16 f32 accumulators).  Block tile = 64 x 128.
// Double-buffered over K (two 32-wide stages in distinct registers) so the
// next stage's global_load_b128s overlap the current stage's WMMAs.
// REQUIRES: K % 64 == 0, M % 64 == 0, N % 128 == 0.
// act: 0 = identity, 1 = softplus.  bias may be null.
// ---------------------------------------------------------------------------
#define LOAD_STAGE(A_, B0_, B1_, B2_, B3_, kof)                                \
  do {                                                                         \
    const unsigned short* pa_ = pa + (kof);                                    \
    const unsigned short* pb_ = pb + (kof);                                    \
    A_.f[0]  = *(const f32x4*)(pa_);                                           \
    A_.f[1]  = *(const f32x4*)(pa_ + 16);                                      \
    B0_.f[0] = *(const f32x4*)(pb_);                                           \
    B0_.f[1] = *(const f32x4*)(pb_ + 16);                                      \
    B1_.f[0] = *(const f32x4*)(pb_ + fs);                                      \
    B1_.f[1] = *(const f32x4*)(pb_ + fs + 16);                                 \
    B2_.f[0] = *(const f32x4*)(pb_ + 2 * fs);                                  \
    B2_.f[1] = *(const f32x4*)(pb_ + 2 * fs + 16);                             \
    B3_.f[0] = *(const f32x4*)(pb_ + 3 * fs);                                  \
    B3_.f[1] = *(const f32x4*)(pb_ + 3 * fs + 16);                             \
  } while (0)

#define WMMA_STAGE(A_, B0_, B1_, B2_, B3_)                                     \
  do {                                                                         \
    acc0 = __builtin_amdgcn_wmma_f32_16x16x32_bf16(false, A_.v, false, B0_.v,  \
                                                   (short)0, acc0, false, false); \
    acc1 = __builtin_amdgcn_wmma_f32_16x16x32_bf16(false, A_.v, false, B1_.v,  \
                                                   (short)0, acc1, false, false); \
    acc2 = __builtin_amdgcn_wmma_f32_16x16x32_bf16(false, A_.v, false, B2_.v,  \
                                                   (short)0, acc2, false, false); \
    acc3 = __builtin_amdgcn_wmma_f32_16x16x32_bf16(false, A_.v, false, B3_.v,  \
                                                   (short)0, acc3, false, false); \
  } while (0)

__global__ void __launch_bounds__(256)
gemm_bf16_wmma(const unsigned short* __restrict__ A,
               const unsigned short* __restrict__ B,
               float* __restrict__ C,
               int K, int lda, int ldb, int ldc,
               const float* __restrict__ bias, int act) {
  int lane = threadIdx.x & 31;
  int wave = threadIdx.x >> 5;
  int mw = blockIdx.y * 64 + (wave >> 1) * 16;
  int nw = blockIdx.x * 128 + (wave & 1) * 64;
  int hl = lane >> 4;                        // half (0/1)
  int ml = lane & 15;                        // row/col within fragment

  const unsigned short* pa = A + (size_t)(mw + ml) * lda + 8 * hl;
  const unsigned short* pb = B + (size_t)(nw + ml) * ldb + 8 * hl;
  const size_t fs = (size_t)16 * ldb;        // fragment-to-fragment row stride

  v8f acc0 = {}, acc1 = {}, acc2 = {}, acc3 = {};
  FragBF aX, b0X, b1X, b2X, b3X;             // stage X
  FragBF aY, b0Y, b1Y, b2Y, b3Y;             // stage Y

  LOAD_STAGE(aX, b0X, b1X, b2X, b3X, 0);
  int k0 = 0;
  for (; k0 + 64 < K; k0 += 64) {
    LOAD_STAGE(aY, b0Y, b1Y, b2Y, b3Y, k0 + 32);
    WMMA_STAGE(aX, b0X, b1X, b2X, b3X);
    LOAD_STAGE(aX, b0X, b1X, b2X, b3X, k0 + 64);
    WMMA_STAGE(aY, b0Y, b1Y, b2Y, b3Y);
  }
  LOAD_STAGE(aY, b0Y, b1Y, b2Y, b3Y, k0 + 32);
  WMMA_STAGE(aX, b0X, b1X, b2X, b3X);
  WMMA_STAGE(aY, b0Y, b1Y, b2Y, b3Y);

  // Epilogue: C/D layout -> row = mw + r + 8*hl, col = nw + 16*i + ml
  v8f accs[4] = {acc0, acc1, acc2, acc3};
#pragma unroll
  for (int i = 0; i < 4; ++i) {
    int col = nw + 16 * i + ml;
    float bv = bias ? bias[col] : 0.0f;
#pragma unroll
    for (int r = 0; r < 8; ++r) {
      int row = mw + r + 8 * hl;
      float v = accs[i][r] + bv;
      if (act == 1) v = (v > 20.0f) ? v : log1pf(__expf(v));
      C[(size_t)row * ldc + col] = v;
    }
  }
}

// ---------------------------------------------------------------------------
// Depthwise causal conv1d (k=4) + bias + SiLU.  Reads x_in = xz[:, :1536],
// writes x_s as both fp32 (for the scan) and bf16 (for the x_proj GEMM).
// ---------------------------------------------------------------------------
__global__ void conv_silu_kernel(const float* __restrict__ xz,
                                 const float* __restrict__ w,
                                 const float* __restrict__ bias,
                                 float* __restrict__ xs_f32,
                                 unsigned short* __restrict__ xs_bf16) {
  int idx = blockIdx.x * blockDim.x + threadIdx.x;
  if (idx >= N_TOK * D_INNER) return;
  int d = idx % D_INNER;
  int tok = idx / D_INNER;
  int t = tok & (L_SEQ - 1);
  float acc = bias[d];
#pragma unroll
  for (int j = 0; j < D_CONV; ++j) {
    int ti = t - (D_CONV - 1) + j;
    if (ti >= 0)
      acc += w[d * D_CONV + j] *
             xz[(size_t)(tok - (D_CONV - 1) + j) * (2 * D_INNER) + d];
  }
  float s = acc / (1.0f + __expf(-acc));     // SiLU
  xs_f32[idx] = s;
  xs_bf16[idx] = f32_to_bf16(s);
}

// ---------------------------------------------------------------------------
// Selective scan, one LANE per (batch, channel, state): 49152 lanes =
// 1536 wave32s of parallelism.  The h-recurrence critical path per token is
// one v_exp + one FMA; the y = sum_n h*C reduction (4x shfl_xor over the
// 16-lane state group) is off the recurrence path.  Lane n==0 of each group
// applies the D-skip and SiLU(z) gate and stores bf16 for out_proj.
// ---------------------------------------------------------------------------
__global__ void scan_kernel(const float* __restrict__ dt,
                            const float* __restrict__ x_dbl,
                            const float* __restrict__ xs,
                            const float* __restrict__ xz,
                            const float* __restrict__ A_log,
                            const float* __restrict__ D_param,
                            unsigned short* __restrict__ y_out) {
  int tid = blockIdx.x * blockDim.x + threadIdx.x;
  int gid = tid >> 4;                 // (b, d) group
  int n   = tid & 15;                 // state index within group
  if (gid >= B_SZ * D_INNER) return;
  int b = gid / D_INNER;
  int d = gid - b * D_INNER;

  float a_n = -__expf(A_log[d * D_STATE + n]);
  float Dd  = D_param[d];
  float h   = 0.0f;

  const float* dt_p = dt    + (size_t)b * L_SEQ * D_INNER + d;
  const float* xs_p = xs    + (size_t)b * L_SEQ * D_INNER + d;
  const float* z_p  = xz    + (size_t)b * L_SEQ * (2 * D_INNER) + D_INNER + d;
  const float* bc_p = x_dbl + (size_t)b * L_SEQ * XP_LD + DT_RANK + n;
  unsigned short* y_p = y_out + (size_t)b * L_SEQ * D_INNER + d;

  for (int t = 0; t < L_SEQ; ++t) {
    float dtv = dt_p[0];
    float xv  = xs_p[0];
    float Bn  = bc_p[0];
    float Cn  = bc_p[D_STATE];
    float dA  = __expf(dtv * a_n);
    h = dA * h + (dtv * xv) * Bn;     // recurrence critical path
    float p = h * Cn;                 // reduction (off critical path)
    p += __shfl_xor(p, 1, 32);
    p += __shfl_xor(p, 2, 32);
    p += __shfl_xor(p, 4, 32);
    p += __shfl_xor(p, 8, 32);
    if (n == 0) {
      float zv = z_p[0];
      float gate = zv / (1.0f + __expf(-zv));   // SiLU(z)
      y_p[0] = f32_to_bf16((p + xv * Dd) * gate);
    }
    dt_p += D_INNER;
    xs_p += D_INNER;
    z_p  += 2 * D_INNER;
    bc_p += XP_LD;
    y_p  += D_INNER;
  }
}

// ---------------------------------------------------------------------------
extern "C" void kernel_launch(void* const* d_in, const int* in_sizes, int n_in,
                              void* d_out, int out_size, void* d_ws, size_t ws_size,
                              hipStream_t stream) {
  const float* x         = (const float*)d_in[0];
  const float* ln_w      = (const float*)d_in[1];
  const float* ln_b      = (const float*)d_in[2];
  const float* in_proj_w = (const float*)d_in[3];   // [3072, 768]
  const float* conv_w    = (const float*)d_in[4];   // [1536, 1, 4]
  const float* conv_b    = (const float*)d_in[5];
  const float* x_proj_w  = (const float*)d_in[6];   // [80, 1536]
  const float* dt_proj_w = (const float*)d_in[7];   // [1536, 48]
  const float* dt_proj_b = (const float*)d_in[8];
  const float* A_log     = (const float*)d_in[9];   // [1536, 16]
  const float* D_param   = (const float*)d_in[10];
  const float* out_proj_w= (const float*)d_in[11];  // [768, 1536]
  float* out = (float*)d_out;

  char* ws = (char*)d_ws;
  size_t off = 0;
  auto walloc = [&](size_t bytes) -> void* {
    void* p = ws + off;
    off = (off + bytes + 255) & ~(size_t)255;
    return p;
  };

  unsigned short* xn_bf = (unsigned short*)walloc((size_t)N_TOK * D_MODEL * 2);
  unsigned short* w_in  = (unsigned short*)walloc((size_t)2 * D_INNER * D_MODEL * 2);
  float*          xz    = (float*)        walloc((size_t)N_TOK * 2 * D_INNER * 4);
  float*          xs_f  = (float*)        walloc((size_t)N_TOK * D_INNER * 4);
  unsigned short* xs_bf = (unsigned short*)walloc((size_t)N_TOK * D_INNER * 2);
  unsigned short* w_xp  = (unsigned short*)walloc((size_t)XP_LD * D_INNER * 2);
  float*          x_dbl = (float*)        walloc((size_t)N_TOK * XP_LD * 4);
  unsigned short* dt_lr = (unsigned short*)walloc((size_t)N_TOK * DT_RANK_P * 2);
  unsigned short* w_dt  = (unsigned short*)walloc((size_t)D_INNER * DT_RANK_P * 2);
  float*          dtbuf = (float*)        walloc((size_t)N_TOK * D_INNER * 4);
  unsigned short* y_bf  = (unsigned short*)walloc((size_t)N_TOK * D_INNER * 2);
  unsigned short* w_out = (unsigned short*)walloc((size_t)D_MODEL * D_INNER * 2);

  const int TB = 256;
  auto cgrid = [](int n) { return (n + 255) / 256; };

  // Weight conversions: fp32 -> bf16; pad x_proj_w rows 80->128 and
  // dt_proj_w cols 48->64 with zeros so every GEMM is guard-free.
  pad_convert_bf16<<<cgrid(2 * D_INNER * D_MODEL), TB, 0, stream>>>(
      in_proj_w, w_in, 2 * D_INNER, 2 * D_INNER, D_MODEL, D_MODEL, D_MODEL);
  pad_convert_bf16<<<cgrid(XP_LD * D_INNER), TB, 0, stream>>>(
      x_proj_w, w_xp, XP_LD, XPROJ_OUT, D_INNER, D_INNER, D_INNER);
  pad_convert_bf16<<<cgrid(D_INNER * DT_RANK_P), TB, 0, stream>>>(
      dt_proj_w, w_dt, D_INNER, D_INNER, DT_RANK, DT_RANK, DT_RANK_P);
  pad_convert_bf16<<<cgrid(D_MODEL * D_INNER), TB, 0, stream>>>(
      out_proj_w, w_out, D_MODEL, D_MODEL, D_INNER, D_INNER, D_INNER);

  // 1) LayerNorm -> bf16
  layernorm_kernel<<<N_TOK, 256, 0, stream>>>(x, ln_w, ln_b, xn_bf);

  // 2) in_proj: [4096,768] x [3072,768]^T -> xz fp32 [4096,3072]
  gemm_bf16_wmma<<<dim3(2 * D_INNER / 128, N_TOK / 64), 256, 0, stream>>>(
      xn_bf, w_in, xz, D_MODEL, D_MODEL, D_MODEL, 2 * D_INNER, nullptr, 0);

  // 3) depthwise conv + SiLU
  conv_silu_kernel<<<cgrid(N_TOK * D_INNER), TB, 0, stream>>>(
      xz, conv_w, conv_b, xs_f, xs_bf);

  // 4) x_proj: [4096,1536] x [128,1536]^T -> x_dbl fp32 [4096,128] (cols 80+ = 0)
  gemm_bf16_wmma<<<dim3(XP_LD / 128, N_TOK / 64), 256, 0, stream>>>(
      xs_bf, w_xp, x_dbl, D_INNER, D_INNER, D_INNER, XP_LD, nullptr, 0);

  // 5) dt low-rank -> bf16, padded 48 -> 64
  pad_convert_bf16<<<cgrid(N_TOK * DT_RANK_P), TB, 0, stream>>>(
      x_dbl, dt_lr, N_TOK, N_TOK, DT_RANK, XP_LD, DT_RANK_P);

  // 6) dt GEMM + bias + softplus: [4096,64] x [1536,64]^T -> dt fp32
  gemm_bf16_wmma<<<dim3(D_INNER / 128, N_TOK / 64), 256, 0, stream>>>(
      dt_lr, w_dt, dtbuf, DT_RANK_P, DT_RANK_P, DT_RANK_P, D_INNER,
      dt_proj_b, 1);

  // 7) selective scan + skip + gating -> y bf16 (one lane per state)
  scan_kernel<<<(B_SZ * D_INNER * D_STATE) / 256, 256, 0, stream>>>(
      dtbuf, x_dbl, xs_f, xz, A_log, D_param, y_bf);

  // 8) out_proj: [4096,1536] x [768,1536]^T -> out fp32
  gemm_bf16_wmma<<<dim3(D_MODEL / 128, N_TOK / 64), 256, 0, stream>>>(
      y_bf, w_out, out, D_INNER, D_INNER, D_INNER, D_MODEL, nullptr, 0);
}
